// Transformer_7980049236366
// MI455X (gfx1250) — compile-verified, matrix operand order checked by reading, not run on hardware
//
#include <hip/hip_runtime.h>

// ---------------------------------------------------------------------------
// Model constants
// ---------------------------------------------------------------------------
constexpr int kB  = 8;
constexpr int kS  = 1024;
constexpr int kM  = 80;     // conv in-channels
constexpr int kD  = 512;
constexpr int kH  = 8;
constexpr int kHD = 64;
constexpr int kF  = 2048;
constexpr int kL  = 6;
constexpr int kV  = 10000;
constexpr int kN  = kB * kS;   // 8192 tokens

typedef __attribute__((ext_vector_type(16))) _Float16 v16h;
typedef __attribute__((ext_vector_type(8)))  _Float16 v8h;
typedef __attribute__((ext_vector_type(8)))  float    v8f;

// Build a 16-half A/B fragment from two contiguous 16-byte segments.
// Per CDNA5 ISA (CDNA5 ISA): 16-bit A 16x32 -> lane (0-15): row=lane,
// elems 0..7 = K[koff..koff+7], elems 8..15 = K[16+koff..], koff=(lane>=16)*8.
__device__ inline v16h load_frag(const _Float16* p0, const _Float16* p1) {
  v8h lo = *(const v8h*)p0;
  v8h hi = *(const v8h*)p1;
  v16h r;
#pragma unroll
  for (int i = 0; i < 8; ++i) { r[i] = lo[i]; r[8 + i] = hi[i]; }
  return r;
}

__device__ inline v8f zero_v8f() {
  v8f z;
#pragma unroll
  for (int i = 0; i < 8; ++i) z[i] = 0.f;
  return z;
}

__device__ inline v8f wmma_f16(v16h a, v16h b, v8f c) {
  return __builtin_amdgcn_wmma_f32_16x16x32_f16(false, a, false, b, (short)0, c,
                                                false, false);
}

// CDNA5 async global->LDS copy (CDNA5 ISA). 16 bytes per active lane,
// tracked by ASYNCcnt. LDS operand: wave-relative byte offset (= low 32 bits of
// the generic shared-memory address per the LDS aperture rule). Inline asm is
// used (portable across ROCm-7.2 / clang-23 builtin arity differences).
__device__ inline void async_load_b128(unsigned lds_byte_off, const void* gaddr) {
  asm volatile("global_load_async_to_lds_b128 %0, %1, off"
               :
               : "v"(lds_byte_off), "v"((unsigned long long)(size_t)gaddr)
               : "memory");
}
__device__ inline void wait_async0() {
  asm volatile("s_wait_asynccnt 0" ::: "memory");
}

// ---------------------------------------------------------------------------
// f32 -> f16 conversion (weights, done once per launch)
// ---------------------------------------------------------------------------
__global__ void cvt_f16_kernel(const float* __restrict__ src,
                               _Float16* __restrict__ dst, long n) {
  long i = (long)blockIdx.x * blockDim.x + threadIdx.x;
  long st = (long)gridDim.x * blockDim.x;
  for (; i < n; i += st) dst[i] = (_Float16)src[i];
}

// ---------------------------------------------------------------------------
// Conv1d stem (k=9, pad 4) + *sqrt(D) + sinusoidal positional embedding.
// grid: (B * S/64), block: 256. Writes h f32 and h f16 copies [N, D].
// ---------------------------------------------------------------------------
__global__ void conv_pos_kernel(const float* __restrict__ x,
                                const float* __restrict__ cw,
                                const float* __restrict__ cb,
                                float* __restrict__ hOut,
                                _Float16* __restrict__ h16Out) {
  __shared__ __align__(16) float xs[kM * 72];  // 80 x (64+8) slab
  const int tid  = threadIdx.x;
  const int b_   = blockIdx.x / (kS / 64);
  const int sblk = blockIdx.x % (kS / 64);

  for (int e = tid; e < kM * 72; e += 256) {
    int m = e / 72, sl = e % 72;
    int gs = sblk * 64 + sl - 4;
    xs[e] = (gs >= 0 && gs < kS) ? x[((size_t)b_ * kM + m) * kS + gs] : 0.f;
  }
  __syncthreads();

  const float loginc = logf(10000.f) / 255.f;  // channels/2 - 1 = 255
#pragma unroll 1
  for (int dh = 0; dh < 2; ++dh) {
    const int dd = tid * 2 + dh;  // 0..511
    float acc[64];
#pragma unroll
    for (int si = 0; si < 64; ++si) acc[si] = 0.f;
    const float* wp = cw + (size_t)dd * kM * 9;
#pragma unroll 1
    for (int m = 0; m < kM; ++m) {
#pragma unroll 1
      for (int k = 0; k < 9; ++k) {
        const float wv = wp[m * 9 + k];
        const float* xr = &xs[m * 72 + k];
#pragma unroll
        for (int si = 0; si < 64; ++si) acc[si] = fmaf(xr[si], wv, acc[si]);
      }
    }
    const float bb  = cb[dd];
    const int   j   = (dd < 256) ? dd : dd - 256;
    const float inv = expf(-loginc * (float)j);
#pragma unroll 1
    for (int si = 0; si < 64; ++si) {
      const int   srow = sblk * 64 + si;
      const float t    = (float)srow * inv;
      const float pe   = (dd < 256) ? sinf(t) : cosf(t);
      const float v    = (acc[si] + bb) * 22.627416997969522f /*sqrt(512)*/ + pe;
      const size_t o   = ((size_t)(b_ * kS + srow)) * kD + dd;
      hOut[o]   = v;
      h16Out[o] = (_Float16)v;
    }
  }
}

// ---------------------------------------------------------------------------
// WMMA GEMM: C[M,N] = act(A[M,K] * W[N,K]^T + bias [+ resid]).
// Block 256 threads = 8 waves; block tile 256(M) x 64(N); wave tile 32 x 64
// (2 A fragments x 4 B fragments -> 8 WMMAs per 32-deep K-step; each B
// fragment feeds two WMMAs). W K-slab (64 cols x 32 halves) is streamed into
// double-buffered LDS with GLOBAL_LOAD_ASYNC_TO_LDS_B128 (ASYNCcnt), one
// workgroup barrier per K-step.
// mode 0: row-major [M,N] outputs. mode 1: f16 scatter to [B,H,S,HD].
// mode 2: f16 scatter to [B,H,HD,S] (transposed V for attention).
// ---------------------------------------------------------------------------
__global__ void gemm_wmma_kernel(const _Float16* __restrict__ A,
                                 const _Float16* __restrict__ W,
                                 const float* __restrict__ bias,
                                 const float* __restrict__ resid,
                                 float* __restrict__ Cf,
                                 _Float16* __restrict__ Ch,
                                 int M, int N, int K, int relu, int mode) {
  __shared__ __align__(16) _Float16 Bs[2][64 * 32];
  const int tid  = threadIdx.x;
  const int wave = tid >> 5;
  const int lane = tid & 31;
  const int r    = lane & 15;
  const int hi   = lane >> 4;
  const int koff = hi * 8;
  const int nbase = blockIdx.x * 64;
  const int mbase = blockIdx.y * 256;
  const int arow0 = mbase + wave * 32 + r;

  v8f acc[8];
#pragma unroll
  for (int t = 0; t < 8; ++t) acc[t] = zero_v8f();

  // Per-thread staging slot: 16 bytes of one W row's K-slab.
  const int c     = tid >> 2;          // 0..63: local output column
  const int chunk = (tid & 3) * 8;     // 0,8,16,24 halves within the 32-k slab
  const int wcol  = nbase + c;
  const bool stg  = (wcol < N);
  const _Float16* wrow = W + (size_t)wcol * K + chunk;
  const unsigned lds0 = (unsigned)(size_t)&Bs[0][c * 32 + chunk];
  const unsigned lds1 = (unsigned)(size_t)&Bs[1][c * 32 + chunk];

  // Prologue: stage k0 = 0 into buffer 0.
  if (stg) async_load_b128(lds0, wrow);
  wait_async0();
  __syncthreads();

  int cur = 0;
  for (int k0 = 0; k0 < K; k0 += 32) {
    // Stage next K-slab into the other buffer while computing on this one.
    if (k0 + 32 < K) {
      if (stg) async_load_b128(cur ? lds0 : lds1, wrow + k0 + 32);
      if (k0 + 64 < K) __builtin_prefetch(wrow + k0 + 64, 0, 1);
    }

    const _Float16* ap0 = A + (size_t)arow0 * K + k0;
    const _Float16* ap1 = ap0 + (size_t)16 * K;
    const v16h a0 = load_frag(ap0 + koff, ap0 + 16 + koff);
    const v16h a1 = load_frag(ap1 + koff, ap1 + 16 + koff);
    const _Float16* bbase = &Bs[cur][0];
#pragma unroll
    for (int t = 0; t < 4; ++t) {
      const _Float16* bp = bbase + (t * 16 + r) * 32;
      const v16h b = load_frag(bp + koff, bp + 16 + koff);
      acc[t]     = wmma_f16(a0, b, acc[t]);
      acc[4 + t] = wmma_f16(a1, b, acc[4 + t]);
    }

    wait_async0();     // my async writes into the other buffer have landed
    __syncthreads();   // everyone done reading cur + other buffer published
    cur ^= 1;
  }

  // Epilogue. C layout: elem j -> row (tile_m + j + 8*hi), col (tile_n + r).
#pragma unroll 1
  for (int half = 0; half < 2; ++half) {
#pragma unroll 1
    for (int t = 0; t < 4; ++t) {
      const int col = nbase + t * 16 + r;
      if (col >= N) continue;
      const float bv = bias ? bias[col] : 0.f;
      const v8f av = acc[half * 4 + t];
#pragma unroll
      for (int j = 0; j < 8; ++j) {
        const int row = mbase + wave * 32 + half * 16 + j + 8 * hi;
        float v = av[j] + bv;
        if (resid) v += resid[(size_t)row * N + col];
        if (relu) v = fmaxf(v, 0.f);
        if (Cf) Cf[(size_t)row * N + col] = v;
        if (Ch) {
          if (mode == 0) {
            Ch[(size_t)row * N + col] = (_Float16)v;
          } else {
            const int b_ = row / kS, s_ = row % kS;
            const int h_ = col / kHD, hd = col % kHD;
            size_t o;
            if (mode == 1)
              o = (((size_t)(b_ * kH + h_)) * kS + s_) * kHD + hd;
            else
              o = (((size_t)(b_ * kH + h_)) * kHD + hd) * kS + s_;
            Ch[o] = (_Float16)v;
          }
        }
      }
    }
  }
}

// ---------------------------------------------------------------------------
// Flash-style attention. grid: (B*H, S/64); block 128 (4 waves).
// Q,K: f16 [B,H,S,64]; V: f16 [B,H,64,S] (transposed). Out: f16 [N, D].
// Wave owns 16 query rows, streams 32 keys/step, online softmax.
// ---------------------------------------------------------------------------
__global__ void attn_kernel(const _Float16* __restrict__ Q,
                            const _Float16* __restrict__ Km,
                            const _Float16* __restrict__ Vt,
                            _Float16* __restrict__ Out) {
  __shared__ __align__(16) _Float16 Ps[4][16 * 32];  // per-wave P tile
  const int bh   = blockIdx.x;
  const int tid  = threadIdx.x;
  const int wave = tid >> 5;
  const int lane = tid & 31;
  const int r    = lane & 15;
  const int hi   = lane >> 4;
  const int koff = hi * 8;
  const int q0   = blockIdx.y * 64 + wave * 16;

  const _Float16* Qb = Q  + (size_t)bh * kS * kHD;
  const _Float16* Kb = Km + (size_t)bh * kS * kHD;
  const _Float16* Vb = Vt + (size_t)bh * kHD * kS;

  const _Float16* qp = Qb + (size_t)(q0 + r) * kHD;
  const v16h qf0 = load_frag(qp + koff,      qp + 16 + koff);
  const v16h qf1 = load_frag(qp + 32 + koff, qp + 48 + koff);

  float mrow[8], lrow[8];
#pragma unroll
  for (int j = 0; j < 8; ++j) { mrow[j] = -3.0e38f; lrow[j] = 0.f; }
  v8f o[4];
#pragma unroll
  for (int t = 0; t < 4; ++t) o[t] = zero_v8f();

  for (int kt = 0; kt < kS; kt += 32) {
    // scores: two 16x16 tiles (keys kt..+15 and kt+16..+31), K-dim = 64
    const _Float16* kp0 = Kb + (size_t)(kt + r) * kHD;
    const _Float16* kp1 = Kb + (size_t)(kt + 16 + r) * kHD;
    v8f s0 = zero_v8f(), s1 = zero_v8f();
    s0 = wmma_f16(qf0, load_frag(kp0 + koff,      kp0 + 16 + koff), s0);
    s0 = wmma_f16(qf1, load_frag(kp0 + 32 + koff, kp0 + 48 + koff), s0);
    s1 = wmma_f16(qf0, load_frag(kp1 + koff,      kp1 + 16 + koff), s1);
    s1 = wmma_f16(qf1, load_frag(kp1 + 32 + koff, kp1 + 48 + koff), s1);

    // online softmax per row (rows striped across the 16 lanes of each half)
#pragma unroll
    for (int j = 0; j < 8; ++j) {
      const float a0 = s0[j] * 0.125f;  // 1/sqrt(64)
      const float a1 = s1[j] * 0.125f;
      float mx = fmaxf(a0, a1);
#pragma unroll
      for (int off = 8; off >= 1; off >>= 1)
        mx = fmaxf(mx, __shfl_xor(mx, off, 16));
      const float mnew  = fmaxf(mrow[j], mx);
      const float scale = expf(mrow[j] - mnew);
      const float p0 = expf(a0 - mnew);
      const float p1 = expf(a1 - mnew);
      float rs = p0 + p1;
#pragma unroll
      for (int off = 8; off >= 1; off >>= 1) rs += __shfl_xor(rs, off, 16);
      lrow[j] = lrow[j] * scale + rs;
      mrow[j] = mnew;
#pragma unroll
      for (int t = 0; t < 4; ++t) o[t][j] *= scale;
      const int prow = j + 8 * hi;
      Ps[wave][prow * 32 + r]      = (_Float16)p0;
      Ps[wave][prow * 32 + 16 + r] = (_Float16)p1;
    }

    // reload P as an A fragment (wave-internal LDS bounce; lockstep wave32)
    const _Float16* pp = &Ps[wave][r * 32];
    const v16h pf = load_frag(pp + koff, pp + 16 + koff);

    // O += P(16x32) * V(32keys x 64hd); Vt gives contiguous B-frag loads
#pragma unroll
    for (int t = 0; t < 4; ++t) {
      const _Float16* vp = Vb + (size_t)(t * 16 + r) * kS + kt;
      const v16h vf = load_frag(vp + koff, vp + 16 + koff);
      o[t] = wmma_f16(pf, vf, o[t]);
    }
  }

  const int h_ = bh % kH;
  const int b_ = bh / kH;
#pragma unroll 1
  for (int t = 0; t < 4; ++t) {
#pragma unroll
    for (int j = 0; j < 8; ++j) {
      const int   s_  = q0 + j + 8 * hi;
      const int   col = h_ * kHD + t * 16 + r;
      const float v   = o[t][j] / lrow[j];
      Out[((size_t)(b_ * kS + s_)) * kD + col] = (_Float16)v;
    }
  }
}

// ---------------------------------------------------------------------------
// LayerNorm over D=512; one row per block (128 threads, 4 elems/thread).
// Writes f32 + f16 copies.
// ---------------------------------------------------------------------------
__global__ void ln_kernel(const float* __restrict__ X,
                          const float* __restrict__ g,
                          const float* __restrict__ b,
                          float* __restrict__ Y,
                          _Float16* __restrict__ Yh) {
  __shared__ float sd[128];
  const int row = blockIdx.x;
  const int tid = threadIdx.x;
  const float* xr = X + (size_t)row * kD;

  float vals[4];
  float s = 0.f;
#pragma unroll
  for (int i = 0; i < 4; ++i) { vals[i] = xr[tid + i * 128]; s += vals[i]; }
  sd[tid] = s; __syncthreads();
  for (int st = 64; st > 0; st >>= 1) {
    if (tid < st) sd[tid] += sd[tid + st];
    __syncthreads();
  }
  const float mean = sd[0] * (1.f / kD);
  __syncthreads();

  float sq = 0.f;
#pragma unroll
  for (int i = 0; i < 4; ++i) { float d = vals[i] - mean; sq += d * d; }
  sd[tid] = sq; __syncthreads();
  for (int st = 64; st > 0; st >>= 1) {
    if (tid < st) sd[tid] += sd[tid + st];
    __syncthreads();
  }
  const float rstd = rsqrtf(sd[0] * (1.f / kD) + 1e-5f);

#pragma unroll
  for (int i = 0; i < 4; ++i) {
    const int col = tid + i * 128;
    const float y = g[col] * (vals[i] - mean) * rstd + b[col];
    Y[(size_t)row * kD + col]  = y;
    Yh[(size_t)row * kD + col] = (_Float16)y;
  }
}

// ---------------------------------------------------------------------------
// Host orchestration
// ---------------------------------------------------------------------------
static inline void* ws_alloc(char*& p, size_t bytes) {
  void* r = p;
  p += (bytes + 255) & ~(size_t)255;
  return r;
}

extern "C" void kernel_launch(void* const* d_in, const int* in_sizes, int n_in,
                              void* d_out, int out_size, void* d_ws, size_t ws_size,
                              hipStream_t stream) {
  const float* x      = (const float*)d_in[0];
  const float* conv_w = (const float*)d_in[1];
  const float* conv_b = (const float*)d_in[2];
  const float* Wq = (const float*)d_in[3];  const float* bq = (const float*)d_in[4];
  const float* Wk = (const float*)d_in[5];  const float* bk = (const float*)d_in[6];
  const float* Wv = (const float*)d_in[7];  const float* bv = (const float*)d_in[8];
  const float* Wo = (const float*)d_in[9];  const float* bo = (const float*)d_in[10];
  const float* g1 = (const float*)d_in[11]; const float* be1 = (const float*)d_in[12];
  const float* W1 = (const float*)d_in[13]; const float* b1 = (const float*)d_in[14];
  const float* W2 = (const float*)d_in[15]; const float* b2 = (const float*)d_in[16];
  const float* g2 = (const float*)d_in[17]; const float* be2 = (const float*)d_in[18];
  const float* out_w = (const float*)d_in[19];
  const float* out_b = (const float*)d_in[20];

  char* wp = (char*)d_ws;
  const size_t DD = (size_t)kD * kD, FD = (size_t)kF * kD;
  _Float16* wq16   = (_Float16*)ws_alloc(wp, kL * DD * 2);
  _Float16* wk16   = (_Float16*)ws_alloc(wp, kL * DD * 2);
  _Float16* wv16   = (_Float16*)ws_alloc(wp, kL * DD * 2);
  _Float16* wo16   = (_Float16*)ws_alloc(wp, kL * DD * 2);
  _Float16* w116   = (_Float16*)ws_alloc(wp, kL * FD * 2);
  _Float16* w216   = (_Float16*)ws_alloc(wp, kL * FD * 2);
  _Float16* outw16 = (_Float16*)ws_alloc(wp, (size_t)kV * kD * 2);
  float*    h      = (float*)   ws_alloc(wp, (size_t)kN * kD * 4);
  _Float16* h16    = (_Float16*)ws_alloc(wp, (size_t)kN * kD * 2);
  _Float16* q16    = (_Float16*)ws_alloc(wp, (size_t)kN * kD * 2);
  _Float16* k16    = (_Float16*)ws_alloc(wp, (size_t)kN * kD * 2);
  _Float16* v16    = (_Float16*)ws_alloc(wp, (size_t)kN * kD * 2);
  _Float16* at16   = (_Float16*)ws_alloc(wp, (size_t)kN * kD * 2);
  float*    t1     = (float*)   ws_alloc(wp, (size_t)kN * kD * 4);
  float*    h1     = (float*)   ws_alloc(wp, (size_t)kN * kD * 4);
  _Float16* h116   = (_Float16*)ws_alloc(wp, (size_t)kN * kD * 2);
  _Float16* ff16   = (_Float16*)ws_alloc(wp, (size_t)kN * kF * 2);
  float*    t2     = (float*)   ws_alloc(wp, (size_t)kN * kD * 4);

  // 1) weights -> f16
  auto cvt = [&](const float* s, _Float16* d, long n) {
    cvt_f16_kernel<<<1024, 256, 0, stream>>>(s, d, n);
  };
  cvt(Wq, wq16, kL * DD); cvt(Wk, wk16, kL * DD);
  cvt(Wv, wv16, kL * DD); cvt(Wo, wo16, kL * DD);
  cvt(W1, w116, kL * FD); cvt(W2, w216, kL * FD);
  cvt(out_w, outw16, (size_t)kV * kD);

  // 2) conv stem + positional embedding
  conv_pos_kernel<<<kB * (kS / 64), 256, 0, stream>>>(x, conv_w, conv_b, h, h16);

  auto gemm = [&](const _Float16* A, const _Float16* W, const float* bias,
                  const float* resid, float* Cf, _Float16* Ch,
                  int M, int N, int K, int relu, int mode) {
    dim3 grid((N + 63) / 64, M / 256);
    gemm_wmma_kernel<<<grid, 256, 0, stream>>>(A, W, bias, resid, Cf, Ch,
                                               M, N, K, relu, mode);
  };

  // 3) transformer layers
  for (int l = 0; l < kL; ++l) {
    gemm(h16, wq16 + l * DD, bq + l * kD, nullptr, nullptr, q16, kN, kD, kD, 0, 1);
    gemm(h16, wk16 + l * DD, bk + l * kD, nullptr, nullptr, k16, kN, kD, kD, 0, 1);
    gemm(h16, wv16 + l * DD, bv + l * kD, nullptr, nullptr, v16, kN, kD, kD, 0, 2);

    attn_kernel<<<dim3(kB * kH, kS / 64), 128, 0, stream>>>(q16, k16, v16, at16);

    gemm(at16, wo16 + l * DD, bo + l * kD, h, t1, nullptr, kN, kD, kD, 0, 0);
    ln_kernel<<<kN, 128, 0, stream>>>(t1, g1 + l * kD, be1 + l * kD, h1, h116);

    gemm(h116, w116 + l * FD, b1 + l * kF, nullptr, nullptr, ff16, kN, kF, kD, 1, 0);
    gemm(ff16, w216 + l * FD, b2 + l * kD, h1, t2, nullptr, kN, kD, kF, 0, 0);
    ln_kernel<<<kN, 128, 0, stream>>>(t2, g2 + l * kD, be2 + l * kD, h, h16);
  }

  // 4) vocab projection -> d_out (f32 [N, V])
  gemm(h16, outw16, out_b, nullptr, (float*)d_out, nullptr, kN, kV, kD, 0, 0);
}
